// MorphingHardAttention_27608049779484
// MI455X (gfx1250) — matrix operand order repaired and checked by reading, not compiled, not run on hardware
//
#include <hip/hip_runtime.h>
#include <hip/hip_bf16.h>
#include <math.h>

typedef __attribute__((ext_vector_type(16))) _Float16 v16h;
typedef __attribute__((ext_vector_type(8)))  _Float16 v8h;
typedef __attribute__((ext_vector_type(4)))  _Float16 v4h;
typedef __attribute__((ext_vector_type(8)))  float    v8f;

#define NNODES 16000
#define NDEG   24
#define NEDGES (NNODES * NDEG)
#define HD     128

// ---------------------------------------------------------------------------
// pack 16 fp32 values (4 x float4) into a v16h fragment register
// ---------------------------------------------------------------------------
__device__ __forceinline__ v16h pack16(float4 f0, float4 f1, float4 f2, float4 f3) {
    v16h h;
    h[0]  = (_Float16)f0.x; h[1]  = (_Float16)f0.y; h[2]  = (_Float16)f0.z; h[3]  = (_Float16)f0.w;
    h[4]  = (_Float16)f1.x; h[5]  = (_Float16)f1.y; h[6]  = (_Float16)f1.z; h[7]  = (_Float16)f1.w;
    h[8]  = (_Float16)f2.x; h[9]  = (_Float16)f2.y; h[10] = (_Float16)f2.z; h[11] = (_Float16)f2.w;
    h[12] = (_Float16)f3.x; h[13] = (_Float16)f3.y; h[14] = (_Float16)f3.z; h[15] = (_Float16)f3.w;
    return h;
}

// Build the 4 A-fragments (K chunks of 32) for one 16-row block.
// A layout (16-bit A 16x32): lane=row, lane-half selects
// K {0-15} vs {16-31} within each chunk (contiguous, matches B).
__device__ __forceinline__ void load_afrags(const float* __restrict__ X,
                                            int row, int laneHi, v16h afrag[4]) {
    const float4* Xr = (const float4*)(X + (size_t)row * HD);
    const int koff = laneHi << 4;
#pragma unroll
    for (int c = 0; c < 4; ++c) {
        const int kb = c * 32 + koff;
        float4 f0 = Xr[(kb >> 2) + 0];
        float4 f1 = Xr[(kb >> 2) + 1];
        float4 f2 = Xr[((kb + 8) >> 2) + 0];
        float4 f3 = Xr[((kb + 8) >> 2) + 1];
        afrag[c] = pack16(f0, f1, f2, f3);
    }
}

// One full 16x128 projection (8 col tiles x 4 K-chunks of WMMA), f32 store.
// Called with concrete kernarg pointers so loads/stores stay GLOBAL (not FLAT).
__device__ __forceinline__ void proj_store_f32(const v16h afrag[4],
                                               const _Float16* __restrict__ Wp,
                                               float* __restrict__ Out,
                                               int rowBlock, int laneHi, int col) {
#pragma unroll 1
    for (int t = 0; t < 8; ++t) {
        const v16h* Wr = (const v16h*)(Wp + (size_t)(t * 16 + col) * HD);
        v8f acc = {};
#pragma unroll
        for (int c = 0; c < 4; ++c) {
            v16h bfrag = Wr[2 * c + laneHi];   // 16 halfs @ K = 32c + 16*laneHi
            acc = __builtin_amdgcn_wmma_f32_16x16x32_f16(
                false, afrag[c], false, bfrag, (short)0, acc, false, false);
        }
#pragma unroll
        for (int r = 0; r < 8; ++r) {
            const int orow = rowBlock * 16 + r + (laneHi << 3);
            Out[(size_t)orow * HD + t * 16 + col] = acc[r];
        }
    }
}

// Same, but epilogue adds a gathered fp32 row and stores f16.
__device__ __forceinline__ void proj_gather_store_f16(const v16h afrag[4],
                                                      const _Float16* __restrict__ Wp,
                                                      const float* __restrict__ Base,
                                                      const int gidx[8],
                                                      _Float16* __restrict__ Out,
                                                      int rowBlock, int laneHi, int col) {
#pragma unroll 1
    for (int t = 0; t < 8; ++t) {
        const v16h* Wr = (const v16h*)(Wp + (size_t)(t * 16 + col) * HD);
        v8f acc = {};
#pragma unroll
        for (int c = 0; c < 4; ++c) {
            v16h bfrag = Wr[2 * c + laneHi];
            acc = __builtin_amdgcn_wmma_f32_16x16x32_f16(
                false, afrag[c], false, bfrag, (short)0, acc, false, false);
        }
#pragma unroll
        for (int r = 0; r < 8; ++r) {
            const int orow = rowBlock * 16 + r + (laneHi << 3);
            const int ocol = t * 16 + col;
            float v = acc[r] + Base[(size_t)gidx[r] * HD + ocol];
            Out[(size_t)orow * HD + ocol] = (_Float16)v;
        }
    }
}

// ---------------------------------------------------------------------------
// fp32 -> fp16 copy (used once for the 5 weight matrices; L2-resident after)
// ---------------------------------------------------------------------------
__global__ __launch_bounds__(256)
void f32_to_f16_kernel(const float* __restrict__ src, _Float16* __restrict__ dst, int n) {
    int i = blockIdx.x * blockDim.x + threadIdx.x;
    if (i < n) dst[i] = (_Float16)src[i];
}

// ---------------------------------------------------------------------------
// Node projections:  Q/Kn/Vn[rows,128] = X @ W{q,k,v}^T
// One wave = one 16-row block; A-fragments built ONCE, reused for 3 weights.
// ---------------------------------------------------------------------------
__global__ __launch_bounds__(256)
void node_proj_kernel(const float* __restrict__ X,
                      const _Float16* __restrict__ W16, // [3][128,128] f16
                      float* __restrict__ Q,
                      float* __restrict__ Kn,
                      float* __restrict__ Vn,
                      int rows)
{
    const int lane     = threadIdx.x & 31;
    const int wave     = threadIdx.x >> 5;
    const int rowBlock = blockIdx.x * 8 + wave;
    if (rowBlock * 16 >= rows) return;

    const int laneHi = lane >> 4;
    const int row    = rowBlock * 16 + (lane & 15);
    const int col    = lane & 15;

    v16h afrag[4];
    load_afrags(X, row, laneHi, afrag);

    const size_t WMAT = (size_t)HD * HD;
    proj_store_f32(afrag, W16 + 0 * WMAT, Q,  rowBlock, laneHi, col);
    proj_store_f32(afrag, W16 + 1 * WMAT, Kn, rowBlock, laneHi, col);
    proj_store_f32(afrag, W16 + 2 * WMAT, Vn, rowBlock, laneHi, col);
}

// ---------------------------------------------------------------------------
// Edge projections:  Ke/Ve[rows,128] = (X @ W{ek,ev}^T + {Kn,Vn}[src]) -> f16
// A-fragments built once per row block, reused for both weights.  Gather rows
// are L2-resident (8MB each); prefetch them ahead of the epilogue.
// ---------------------------------------------------------------------------
__global__ __launch_bounds__(256)
void edge_proj_kernel(const float* __restrict__ X,
                      const _Float16* __restrict__ W16, // [2][128,128] f16 (Wek, Wev)
                      const float* __restrict__ Kn,
                      const float* __restrict__ Vn,
                      const int*   __restrict__ src_idx,
                      _Float16* __restrict__ Ke,
                      _Float16* __restrict__ Ve,
                      int rows)
{
    const int lane     = threadIdx.x & 31;
    const int wave     = threadIdx.x >> 5;
    const int rowBlock = blockIdx.x * 8 + wave;
    if (rowBlock * 16 >= rows) return;

    const int laneHi = lane >> 4;
    const int row    = rowBlock * 16 + (lane & 15);
    const int col    = lane & 15;

    v16h afrag[4];
    load_afrags(X, row, laneHi, afrag);

    int gidx[8];
#pragma unroll
    for (int r = 0; r < 8; ++r) {
        gidx[r] = src_idx[rowBlock * 16 + r + (laneHi << 3)];
        __builtin_prefetch(Kn + (size_t)gidx[r] * HD + col, 0, 3);
        __builtin_prefetch(Vn + (size_t)gidx[r] * HD + col, 0, 3);
    }

    const size_t WMAT = (size_t)HD * HD;
    proj_gather_store_f16(afrag, W16 + 0 * WMAT, Kn, gidx, Ke, rowBlock, laneHi, col);
    proj_gather_store_f16(afrag, W16 + 1 * WMAT, Vn, gidx, Ve, rowBlock, laneHi, col);
}

// ---------------------------------------------------------------------------
// Attention: one wave32 per node.  25 valid slots (self + 24 in-edges; the 7
// padded slots are masked -> provably zero probability).  Lane m holds logit
// m; bitonic sort + wave scans give the entmax-1.5 / sparsemax thresholds;
// forward value of the straight-through attn = hard mean over selected.
// Edge residual is fused: this wave owns edge rows [n*24, n*24+24).
// ---------------------------------------------------------------------------
__global__ __launch_bounds__(256)
void morph_attn_kernel(const float* __restrict__ Q,
                       const float* __restrict__ Kn,
                       const float* __restrict__ Vn,
                       const _Float16* __restrict__ Ke,
                       const _Float16* __restrict__ Ve,
                       const float* __restrict__ node_states,
                       const float* __restrict__ edge_states,
                       const float* __restrict__ alpha_param,
                       float* __restrict__ node_fts,   // d_out[0 : N*H)
                       float* __restrict__ edge_fts,   // d_out[N*H : N*H+E*H)
                       int n_nodes)
{
    const int lane = threadIdx.x & 31;
    const int node = blockIdx.x * 8 + (threadIdx.x >> 5);
    if (node >= n_nodes) return;

    const int   NSLOT = NDEG + 1;             // 25 valid of 32
    const float scale = 0.08838834764831845f; // 1/sqrt(128)

    // ---- logits: lane m computes dot(Q[n], K_m) --------------------------
    float logit = -1e30f;
    const float4* q4 = (const float4*)(Q + (size_t)node * HD);
    if (lane == 0) {
        const float4* k4 = (const float4*)(Kn + (size_t)node * HD);
        float s = 0.f;
#pragma unroll 8
        for (int i = 0; i < HD / 4; ++i) {
            float4 a = q4[i], b = k4[i];
            s += a.x * b.x + a.y * b.y + a.z * b.z + a.w * b.w;
        }
        logit = s * scale;
    } else if (lane < NSLOT) {
        const v8h* kr = (const v8h*)(Ke + ((size_t)node * NDEG + (lane - 1)) * HD);
        float s = 0.f;
#pragma unroll 4
        for (int i = 0; i < HD / 8; ++i) {
            v8h k8 = kr[i];
            float4 a = q4[2 * i], b = q4[2 * i + 1];
            s += a.x * (float)k8[0] + a.y * (float)k8[1] + a.z * (float)k8[2] + a.w * (float)k8[3]
               + b.x * (float)k8[4] + b.y * (float)k8[5] + b.z * (float)k8[6] + b.w * (float)k8[7];
        }
        logit = s * scale;
    }

    // ---- softmax pieces ---------------------------------------------------
    float mx = logit;
#pragma unroll
    for (int o = 16; o; o >>= 1) mx = fmaxf(mx, __shfl_xor(mx, o, 32));
    float ex = (lane < NSLOT) ? __expf(logit - mx) : 0.f;
    float Zs = ex;
#pragma unroll
    for (int o = 16; o; o >>= 1) Zs += __shfl_xor(Zs, o, 32);
    const float psoft = ex / Zs;

    // ---- bitonic sort (descending) across the wave ------------------------
    float z = logit;
#pragma unroll
    for (int k = 2; k <= 32; k <<= 1) {
#pragma unroll
        for (int j = k >> 1; j > 0; j >>= 1) {
            float o  = __shfl_xor(z, j, 32);
            float mn = fminf(z, o), mxv = fmaxf(z, o);
            bool asc = ((lane & k) == 0);
            bool low = ((lane & j) == 0);
            z = (asc == low) ? mxv : mn;     // descending
        }
    }

    // ---- inclusive scans of z and z^2 ------------------------------------
    float cs = z, cs2 = z * z;
#pragma unroll
    for (int o = 1; o < 32; o <<= 1) {
        float t1 = __shfl_up(cs, o, 32);
        float t2 = __shfl_up(cs2, o, 32);
        if (lane >= o) { cs += t1; cs2 += t2; }
    }

    const float kf = (float)(lane + 1);
    // entmax-1.5 threshold
    float mean_z  = cs / kf;
    float mean_z2 = cs2 / kf;
    float discr   = fmaxf(mean_z * mean_z - mean_z2 + 1.f / kf, 0.f);
    float tau_c   = mean_z - sqrtf(discr);
    unsigned long long bal = __ballot((lane < NSLOT) && (z > tau_c));
    int   kidx  = (int)__popcll(bal);
    float tau15 = __shfl(tau_c, kidx - 1, 32);
    // sparsemax threshold
    bal = __ballot((lane < NSLOT) && (kf * z > cs - 1.f));
    int   kidx2 = (int)__popcll(bal);
    float csk   = __shfl(cs, kidx2 - 1, 32);
    float tausp = (csk - 1.f) / (float)kidx2;

    // ---- morphing mix -----------------------------------------------------
    float p = 0.f;
    if (lane < NSLOT) {
        float r15 = fmaxf(logit - tau15, 0.f);
        float p15 = r15 * r15;
        float psp = fmaxf(logit - tausp, 0.f);
        float u   = 1.f / (1.f + __expf(-alpha_param[0]));
        if (u <= 0.5f) { float w = 2.f * u;       p = (1.f - w) * psoft + w * p15; }
        else           { float w = 2.f * u - 1.f; p = (1.f - w) * p15   + w * psp; }
    }

    // ---- hard (straight-through forward) weights -------------------------
    bool sel = (lane < NSLOT) && (p > 1e-6f);
    bal = __ballot(sel);
    int   nsel = (int)__popcll(bal);
    float wgt  = sel ? 1.f / (float)nsel : 0.f;

    // ---- output = sum_m w_m * V_m ; lanes parallel over H ----------------
    float4 acc = make_float4(0.f, 0.f, 0.f, 0.f);
    {
        float w0 = __shfl(wgt, 0, 32);
        if (w0 != 0.f) {
            float4 v = ((const float4*)(Vn + (size_t)node * HD))[lane];
            acc.x += w0 * v.x; acc.y += w0 * v.y; acc.z += w0 * v.z; acc.w += w0 * v.w;
        }
    }
    for (int m = 1; m < NSLOT; ++m) {
        float wm = __shfl(wgt, m, 32);
        if (wm != 0.f) {
            v4h v = ((const v4h*)(Ve + ((size_t)node * NDEG + (m - 1)) * HD))[lane];
            acc.x += wm * (float)v[0]; acc.y += wm * (float)v[1];
            acc.z += wm * (float)v[2]; acc.w += wm * (float)v[3];
        }
    }

    // ---- node residual ----------------------------------------------------
    float4 base = ((const float4*)(node_states + (size_t)node * HD))[lane];
    ((float4*)(node_fts + (size_t)node * HD))[lane] =
        make_float4(base.x + acc.x, base.y + acc.y, base.z + acc.z, base.w + acc.w);

    // ---- fused edge residual: this node's 24 edges are contiguous rows ----
    const float4* es = (const float4*)(edge_states + (size_t)node * NDEG * HD);
    float4*       eo = (float4*)(edge_fts + (size_t)node * NDEG * HD);
#pragma unroll 4
    for (int j = 0; j < NDEG; ++j) {
        float4 a = es[j * (HD / 4) + lane];
        eo[j * (HD / 4) + lane] =
            make_float4(a.x + acc.x, a.y + acc.y, a.z + acc.z, a.w + acc.w);
    }
}

// ---------------------------------------------------------------------------
extern "C" void kernel_launch(void* const* d_in, const int* in_sizes, int n_in,
                              void* d_out, int out_size, void* d_ws, size_t ws_size,
                              hipStream_t stream) {
    const float* node_states = (const float*)d_in[0];
    const float* edge_states = (const float*)d_in[1];
    const int*   src_idx     = (const int*)d_in[2];
    // d_in[3] = dst_idx (structure is fixed-degree sorted: edge e -> node e/24)
    // d_in[4] = training_step (unused)
    const float* Wq  = (const float*)d_in[5];
    const float* Wk  = (const float*)d_in[6];
    const float* Wv  = (const float*)d_in[7];
    const float* Wek = (const float*)d_in[8];
    const float* Wev = (const float*)d_in[9];
    const float* alpha_param = (const float*)d_in[10];

    float* out_node = (float*)d_out;                       // [N,H]
    float* out_edge = (float*)d_out + (size_t)NNODES * HD; // [E,H]

    // ---- workspace layout -------------------------------------------------
    const size_t WMAT = (size_t)HD * HD;                   // 16384
    _Float16* W16 = (_Float16*)d_ws;                       // 5 f16 weight mats
    float* Qb = (float*)(W16 + 5 * WMAT);
    float* Kn = Qb + (size_t)NNODES * HD;
    float* Vn = Kn + (size_t)NNODES * HD;
    _Float16* Ke = (_Float16*)(Vn + (size_t)NNODES * HD);
    _Float16* Ve = Ke + (size_t)NEDGES * HD;

    // ---- 1. weights -> f16 (tiny; stays L2-resident) ----------------------
    const int cvtBlocks = (int)((WMAT + 255) / 256);
    f32_to_f16_kernel<<<cvtBlocks, 256, 0, stream>>>(Wq,  W16 + 0 * WMAT, (int)WMAT);
    f32_to_f16_kernel<<<cvtBlocks, 256, 0, stream>>>(Wk,  W16 + 1 * WMAT, (int)WMAT);
    f32_to_f16_kernel<<<cvtBlocks, 256, 0, stream>>>(Wv,  W16 + 2 * WMAT, (int)WMAT);
    f32_to_f16_kernel<<<cvtBlocks, 256, 0, stream>>>(Wek, W16 + 3 * WMAT, (int)WMAT);
    f32_to_f16_kernel<<<cvtBlocks, 256, 0, stream>>>(Wev, W16 + 4 * WMAT, (int)WMAT);

    // ---- 2. node projections (Q,K,V share A-fragments) --------------------
    const int nodeBlocks = (NNODES / 16 + 7) / 8;  // 125
    node_proj_kernel<<<nodeBlocks, 256, 0, stream>>>(
        node_states, W16, Qb, Kn, Vn, NNODES);

    // ---- 3. edge projections (Ke,Ve share A-fragments; gather fused) ------
    const int edgeBlocks = (NEDGES / 16 + 7) / 8;  // 3000
    edge_proj_kernel<<<edgeBlocks, 256, 0, stream>>>(
        edge_states, W16 + 3 * WMAT, Kn, Vn, src_idx, Ke, Ve, NEDGES);

    // ---- 4. attention + both residuals (one wave per node) ----------------
    morph_attn_kernel<<<(NNODES + 7) / 8, 256, 0, stream>>>(
        Qb, Kn, Vn, Ke, Ve, node_states, edge_states, alpha_param,
        out_node, out_edge, NNODES);
}
